// LSTM_13022340842263
// MI455X (gfx1250) — matrix operand order repaired and checked by reading, not compiled, not run on hardware
//
#include <hip/hip_runtime.h>
#include <hip/hip_bf16.h>
#include <cstdint>
#include <cstddef>

// Problem constants (match reference)
#define T_LEN   2048
#define BATCH   32
#define IN_DIM  128
#define HID     256
#define GATES   1024   // 4*HID
#define CAT     512    // 2*HID
#define OUT_DIM 128
#define M_ROWS  (T_LEN * BATCH)   // 65536

typedef __attribute__((ext_vector_type(16))) _Float16 v16h;
typedef __attribute__((ext_vector_type(8)))  _Float16 v8h;
typedef __attribute__((ext_vector_type(8)))  float    v8f;

typedef unsigned int u32x4 __attribute__((ext_vector_type(4)));
typedef int          i32x4 __attribute__((ext_vector_type(4)));
typedef int          i32x8 __attribute__((ext_vector_type(8)));

// ---------------------------------------------------------------------------
// WMMA helpers (CDNA5 wave32, 16x16x32 f16 -> f32)
// ---------------------------------------------------------------------------
__device__ __forceinline__ v8f wmma_f16(v16h a, v16h b, v8f c) {
  // (neg_a, A, neg_b, B, c_mod, C, reuse_a, reuse_b)
  return __builtin_amdgcn_wmma_f32_16x16x32_f16(false, a, false, b, (short)0, c,
                                                false, false);
}

// A fragment: 16x32 (MxK) tile from row-major X[*, ld], top-left (m0, k0).
// lanes 0-15 (row m=lane): K k0..k0+7 (v0-3), k0+16..k0+23 (v4-7);
// lanes 16-31: K k0+8..15 and k0+24..31. Two b128 loads per lane.
__device__ __forceinline__ v16h load_frag_A(const _Float16* X, int ld, int m0,
                                            int k0) {
  const int lane = threadIdx.x & 31;
  const int m  = m0 + (lane & 15);
  const int kb = k0 + ((lane >> 4) << 3);
  const _Float16* p = X + (size_t)m * ld + kb;
  v8h lo = *(const v8h*)p;
  v8h hi = *(const v8h*)(p + 16);
  v16h a;
#pragma unroll
  for (int i = 0; i < 8; ++i) { a[i] = lo[i]; a[i + 8] = hi[i]; }
  return a;
}

// B fragment: 32x16 (KxN), B = W^T with W row-major [N, ld], tile (k0, n0).
// lane<16: col n=lane holds K k0..k0+7, k0+16..k0+23; lane>=16: K k0+8..15, k0+24..31.
__device__ __forceinline__ v16h load_frag_B(const _Float16* W, int ld, int n0,
                                            int k0) {
  const int lane = threadIdx.x & 31;
  const int n  = n0 + (lane & 15);
  const int kb = k0 + ((lane >> 4) << 3);
  const _Float16* p = W + (size_t)n * ld + kb;
  v8h lo = *(const v8h*)p;
  v8h hi = *(const v8h*)(p + 16);
  v16h b;
#pragma unroll
  for (int i = 0; i < 8; ++i) { b[i] = lo[i]; b[i + 8] = hi[i]; }
  return b;
}

// C/D tile stores: lane<16 -> rows m0+0..7, lane>=16 -> rows m0+8..15,
// column n0+(lane&15).
__device__ __forceinline__ void store_tile_f32(float* Y, int ld, int m0,
                                               int n0, v8f c,
                                               const float* bias) {
  const int lane = threadIdx.x & 31;
  const int n  = n0 + (lane & 15);
  const int mo = (lane >> 4) << 3;
  const float bv = bias ? bias[n] : 0.0f;
#pragma unroll
  for (int r = 0; r < 8; ++r)
    Y[(size_t)(m0 + mo + r) * ld + n] = c[r] + bv;
}

__device__ __forceinline__ void store_tile_f16(_Float16* Y, int ld, int m0,
                                               int n0, v8f c,
                                               const float* bias) {
  const int lane = threadIdx.x & 31;
  const int n  = n0 + (lane & 15);
  const int mo = (lane >> 4) << 3;
  const float bv = bias ? bias[n] : 0.0f;
#pragma unroll
  for (int r = 0; r < 8; ++r)
    Y[(size_t)(m0 + mo + r) * ld + n] = (_Float16)(c[r] + bv);
}

__device__ __forceinline__ float sigmoidf_(float x) {
  return 1.0f / (1.0f + __expf(-x));
}

// ---------------------------------------------------------------------------
// Tensor Data Mover: 2D tile (rows x cols of f16, row stride in elements)
// Global -> LDS, packed contiguously at lds_addr. ISSUE ONLY (no wait); the
// issuing wave must s_wait_tensorcnt(0) before the LDS data is consumed.
// D# packing per CDNA5 ISA ch.8 (group0 128b, group1 256b).
// ---------------------------------------------------------------------------
#if defined(__gfx1250__)
__device__ __forceinline__ void tdm_issue_tile_f16(uint32_t lds_addr,
                                                   const _Float16* gptr,
                                                   int rows, int cols,
                                                   int row_stride_elems,
                                                   int tensor_rows) {
  const uint64_t ga = (uint64_t)(uintptr_t)gptr;
  u32x4 g0;
  g0[0] = 1u;                                   // count=1, user descriptor
  g0[1] = lds_addr;                             // lds_addr[31:0]
  g0[2] = (uint32_t)(ga & 0xFFFFFFFFu);         // global_addr[31:0]
  g0[3] = (uint32_t)((ga >> 32) & 0x1FFFFFFu)   // global_addr[56:32]
          | (2u << 30);                         // type = 2 ("image")
  const uint32_t td0 = (uint32_t)row_stride_elems;  // tensor dim0 length
  const uint32_t td1 = (uint32_t)tensor_rows;       // tensor dim1 length
  const uint64_t s0  = (uint64_t)row_stride_elems;  // dim0 stride (elems)
  i32x8 g1;
  g1[0] = (int)(1u << 16);                      // wg_mask=0, data_size=2B
  g1[1] = (int)((td0 & 0xFFFFu) << 16);         // tensor_dim0[15:0]
  g1[2] = (int)((td0 >> 16) | ((td1 & 0xFFFFu) << 16));
  g1[3] = (int)((td1 >> 16) | ((uint32_t)cols << 16));  // tile_dim0
  g1[4] = (int)(uint32_t)rows;                  // tile_dim1 (tile_dim2=0)
  g1[5] = (int)(uint32_t)(s0 & 0xFFFFFFFFu);    // dim0_stride[31:0]
  g1[6] = (int)(uint32_t)((s0 >> 32) & 0xFFFFu);
  g1[7] = 0;
  const i32x4 gz = {0, 0, 0, 0};
#if defined(__clang_major__) && (__clang_major__ >= 23)
  const i32x8 gz8 = {0, 0, 0, 0, 0, 0, 0, 0};
  __builtin_amdgcn_tensor_load_to_lds(g0, g1, gz, gz, gz8, 0);
#else
  __builtin_amdgcn_tensor_load_to_lds(g0, g1, gz, gz, 0);
#endif
}
#define TDM_WAIT_WAVE0()                                   \
  do {                                                     \
    if (threadIdx.x < 32) __builtin_amdgcn_s_wait_tensorcnt(0); \
  } while (0)
#else
#define TDM_WAIT_WAVE0() do { } while (0)
#endif

// ---------------------------------------------------------------------------
// Small prep kernels
// ---------------------------------------------------------------------------
__global__ void f32_to_f16_kernel(const float* __restrict__ in,
                                  _Float16* __restrict__ out, int n) {
  int i = blockIdx.x * blockDim.x + threadIdx.x;
  if (i < n) out[i] = (_Float16)in[i];
}

// x: [B, IN, T] f32 -> seq: [T, B, IN] f16
__global__ void pack_seq_kernel(const float* __restrict__ x,
                                _Float16* __restrict__ seq) {
  int idx = blockIdx.x * blockDim.x + threadIdx.x;
  if (idx >= T_LEN * BATCH * IN_DIM) return;
  int i = idx % IN_DIM;
  int b = (idx / IN_DIM) % BATCH;
  int t = idx / (IN_DIM * BATCH);
  seq[idx] = (_Float16)x[((size_t)b * IN_DIM + i) * T_LEN + t];
}

// ---------------------------------------------------------------------------
// Input-projection GEMM: Y[M,1024](f16) = X[M,K] @ W[1024,K]^T + bias
// 256 thr = 8 waves (2x4), block tile 64M x 256N, BK=32.
// Double-buffered LDS pipeline: stage buffer p^1 (A: vector loads, W: TDM
// issue) while computing from buffer p; defer s_wait_tensorcnt to after the
// WMMA phase -> one barrier per K-iteration, DMA fully overlapped.
// ---------------------------------------------------------------------------
#define BM 64
#define BN 256
#define BK 32

__global__ __launch_bounds__(256) void gemm_xw_kernel(
    const _Float16* __restrict__ X, const _Float16* __restrict__ W,
    const float* __restrict__ bias, _Float16* __restrict__ Y, int M, int K) {
  __shared__ _Float16 As[2][BM][BK];   // 2 x 4 KB
  __shared__ _Float16 Ws[2][BN][BK];   // 2 x 16 KB

  const int m0 = blockIdx.x * BM;
  const int n0 = blockIdx.y * BN;
  const int wave = threadIdx.x >> 5;   // 0..7
  const int wm = wave & 1;             // 2 wave-rows of 32
  const int wn = wave >> 1;            // 4 wave-cols of 64

  v8f acc[2][4] = {};

  const int arow = threadIdx.x >> 2;        // 0..63
  const int acol = (threadIdx.x & 3) * 8;   // 0,8,16,24

  auto stage = [&](int p, int kb) {
    // A tile (64x32 halves): global_load_b128 -> ds_store_b128
    *(v8h*)&As[p][arow][acol] =
        *(const v8h*)&X[(size_t)(m0 + arow) * K + kb + acol];
    // W tile (256 rows x 32 halves, row stride K)
#if defined(__gfx1250__)
    if (threadIdx.x < 32)
      tdm_issue_tile_f16((uint32_t)(uintptr_t)&Ws[p][0][0],
                         &W[(size_t)n0 * K + kb], BN, BK, K, GATES);
#else
    const int wrow = threadIdx.x;
    const _Float16* src = &W[(size_t)(n0 + wrow) * K + kb];
    *(v8h*)&Ws[p][wrow][0]  = *(const v8h*)&src[0];
    *(v8h*)&Ws[p][wrow][8]  = *(const v8h*)&src[8];
    *(v8h*)&Ws[p][wrow][16] = *(const v8h*)&src[16];
    *(v8h*)&Ws[p][wrow][24] = *(const v8h*)&src[24];
#endif
  };

  stage(0, 0);
  TDM_WAIT_WAVE0();
  __syncthreads();

  int p = 0;
  for (int kb = 0; kb < K; kb += BK) {
    if (kb + BK < K) {
      stage(p ^ 1, kb + BK);  // overlaps with compute below
      if (kb + 2 * BK < K)
        __builtin_prefetch(&X[(size_t)(m0 + arow) * K + kb + 2 * BK + acol],
                           0, 1);
    }
    v16h af0 = load_frag_A(&As[p][0][0], BK, wm * 32 + 0, 0);
    v16h af1 = load_frag_A(&As[p][0][0], BK, wm * 32 + 16, 0);
#pragma unroll
    for (int nt = 0; nt < 4; ++nt) {
      v16h bf = load_frag_B(&Ws[p][0][0], BK, wn * 64 + nt * 16, 0);
      acc[0][nt] = wmma_f16(af0, bf, acc[0][nt]);
      acc[1][nt] = wmma_f16(af1, bf, acc[1][nt]);
    }
    TDM_WAIT_WAVE0();   // next buffer's DMA done (overlapped with WMMAs)
    __syncthreads();    // publishes p^1, retires reads of p
    p ^= 1;
  }

#pragma unroll
  for (int mt = 0; mt < 2; ++mt)
#pragma unroll
    for (int nt = 0; nt < 4; ++nt)
      store_tile_f16(Y, GATES, m0 + wm * 32 + mt * 16, n0 + wn * 64 + nt * 16,
                     acc[mt][nt], bias);
}

// ---------------------------------------------------------------------------
// Recurrent LSTM layer, both directions (blockIdx.x: 0=fwd, 1=bwd).
// 512 threads = 16 waves; wave w owns hidden-column tile jt=w for ALL four
// gates and BOTH batch tiles -> each w_hh row fetched once per step and the
// cell update is fully lane-local (c in registers across all T steps).
// xw[t] (64 KB contiguous) is TDM double-buffered into LDS: the DMA for
// step t+1 is issued before the gate GEMM of step t and waited only after
// the elementwise phase -> its latency is fully hidden.
// LDS: hs 16 KB + xw ping-pong 128 KB = 144 KB (< 320 KB WGP budget).
// ---------------------------------------------------------------------------
__global__ __launch_bounds__(512) void lstm_layer_kernel(
    const _Float16* __restrict__ xw_f, const _Float16* __restrict__ xw_b,
    const _Float16* __restrict__ whh_f, const _Float16* __restrict__ whh_b,
    _Float16* __restrict__ hcat) {
  const int dir = blockIdx.x;
  const _Float16* __restrict__ xw = dir ? xw_b : xw_f;
  const _Float16* __restrict__ whh = dir ? whh_b : whh_f;

  __shared__ _Float16 hs[BATCH][HID];          // 16 KB hidden state
  __shared__ _Float16 xbuf[2][BATCH][GATES];   // 128 KB xw ping-pong

  for (int i = threadIdx.x; i < BATCH * HID; i += blockDim.x)
    (&hs[0][0])[i] = (_Float16)0.0f;

  const int jt = threadIdx.x >> 5;     // wave = j tile, 0..15
  const int lane = threadIdx.x & 31;
  const int n  = lane & 15;
  const int mo = (lane >> 4) << 3;
  const int j = jt * 16 + n;           // this lane's hidden column

  // Stage xw[t] slab (32 rows x 1024 halves, contiguous) into xbuf[buf]
  auto stage_xw = [&](int buf, int t) {
    const _Float16* src = xw + (size_t)t * (BATCH * GATES);
#if defined(__gfx1250__)
    if (threadIdx.x < 32)
      tdm_issue_tile_f16((uint32_t)(uintptr_t)&xbuf[buf][0][0], src, BATCH,
                         GATES, GATES, BATCH * T_LEN);
#else
    for (int i = threadIdx.x; i < (BATCH * GATES) / 8; i += blockDim.x)
      ((v8h*)&xbuf[buf][0][0])[i] = ((const v8h*)src)[i];
#endif
  };

  float c_state[2][8];
#pragma unroll
  for (int mt = 0; mt < 2; ++mt)
#pragma unroll
    for (int r = 0; r < 8; ++r) c_state[mt][r] = 0.0f;

  // Prologue: preload xw for the first timestep
  stage_xw(0, dir ? (T_LEN - 1) : 0);
  TDM_WAIT_WAVE0();
  __syncthreads();

  for (int step = 0; step < T_LEN; ++step) {
    const int t = dir ? (T_LEN - 1 - step) : step;
    const int cur = step & 1;

    if (step + 1 < T_LEN)  // issue DMA for next step; overlaps GEMM below
      stage_xw(cur ^ 1, dir ? (T_LEN - 2 - step) : (step + 1));

    v8f acc[2][4] = {};   // [batch tile][gate i,f,g,o]
#pragma unroll
    for (int kk = 0; kk < HID; kk += 32) {
      v16h a0 = load_frag_A(&hs[0][0], HID, 0, kk);
      v16h a1 = load_frag_A(&hs[0][0], HID, 16, kk);
#pragma unroll
      for (int g = 0; g < 4; ++g) {
        v16h bg = load_frag_B(whh, HID, g * HID + jt * 16, kk);
        acc[0][g] = wmma_f16(a0, bg, acc[0][g]);
        acc[1][g] = wmma_f16(a1, bg, acc[1][g]);
      }
    }
    __syncthreads();  // all waves done reading hs

#pragma unroll
    for (int mt = 0; mt < 2; ++mt) {
#pragma unroll
      for (int r = 0; r < 8; ++r) {
        const int b = mt * 16 + mo + r;
        const _Float16* __restrict__ xr = &xbuf[cur][b][0];
        float gi = acc[mt][0][r] + (float)xr[0 * HID + j];
        float gf = acc[mt][1][r] + (float)xr[1 * HID + j];
        float gg = acc[mt][2][r] + (float)xr[2 * HID + j];
        float go = acc[mt][3][r] + (float)xr[3 * HID + j];
        gi = sigmoidf_(gi);
        gf = sigmoidf_(gf);
        go = sigmoidf_(go);
        gg = tanhf(gg);
        float c = gf * c_state[mt][r] + gi * gg;
        c_state[mt][r] = c;
        float h = go * tanhf(c);
        _Float16 hh = (_Float16)h;
        hs[b][j] = hh;
        hcat[((size_t)t * BATCH + b) * CAT + dir * HID + j] = hh;
      }
    }
    TDM_WAIT_WAVE0();  // xbuf[cur^1] DMA complete (hidden behind this step)
    __syncthreads();   // hs + next xw buffer published
  }
}

// ---------------------------------------------------------------------------
// Final FC: out[32,128] = hlast[32,512] @ fc_w[128,512]^T + fc_b
// ---------------------------------------------------------------------------
__global__ __launch_bounds__(256) void fc_kernel(
    const _Float16* __restrict__ hlast, const _Float16* __restrict__ fcw,
    const float* __restrict__ fcb, float* __restrict__ out) {
  const int wave = threadIdx.x >> 5;   // 0..7 -> N tile
  v8f acc0 = {}, acc1 = {};
#pragma unroll
  for (int kb = 0; kb < CAT; kb += 32) {
    v16h a0 = load_frag_A(hlast, CAT, 0, kb);
    v16h a1 = load_frag_A(hlast, CAT, 16, kb);
    v16h b  = load_frag_B(fcw, CAT, wave * 16, kb);
    acc0 = wmma_f16(a0, b, acc0);
    acc1 = wmma_f16(a1, b, acc1);
  }
  store_tile_f32(out, OUT_DIM, 0,  wave * 16, acc0, fcb);
  store_tile_f32(out, OUT_DIM, 16, wave * 16, acc1, fcb);
}

// ---------------------------------------------------------------------------
// Host orchestration
// ---------------------------------------------------------------------------
extern "C" void kernel_launch(void* const* d_in, const int* in_sizes, int n_in,
                              void* d_out, int out_size, void* d_ws,
                              size_t ws_size, hipStream_t stream) {
  (void)in_sizes; (void)n_in; (void)out_size; (void)ws_size;

  const float* x       = (const float*)d_in[0];
  const float* w_ih_0f = (const float*)d_in[1];
  const float* w_hh_0f = (const float*)d_in[2];
  const float* b_0f    = (const float*)d_in[3];
  const float* w_ih_0b = (const float*)d_in[4];
  const float* w_hh_0b = (const float*)d_in[5];
  const float* b_0b    = (const float*)d_in[6];
  const float* w_ih_1f = (const float*)d_in[7];
  const float* w_hh_1f = (const float*)d_in[8];
  const float* b_1f    = (const float*)d_in[9];
  const float* w_ih_1b = (const float*)d_in[10];
  const float* w_hh_1b = (const float*)d_in[11];
  const float* b_1b    = (const float*)d_in[12];
  const float* fc_w    = (const float*)d_in[13];
  const float* fc_b    = (const float*)d_in[14];
  float* out = (float*)d_out;

  // Workspace carve-up (~405 MB total; deterministic pointer math only).
  uint8_t* ws = (uint8_t*)d_ws;
  size_t off = 0;
  auto take = [&](size_t bytes) -> uint8_t* {
    uint8_t* p = ws + off;
    off += (bytes + 255) & ~(size_t)255;
    return p;
  };
  _Float16* seq0   = (_Float16*)take((size_t)M_ROWS * IN_DIM * 2);   // 16 MB
  _Float16* wih0f  = (_Float16*)take((size_t)GATES * IN_DIM * 2);
  _Float16* whh0f  = (_Float16*)take((size_t)GATES * HID * 2);
  _Float16* wih0b  = (_Float16*)take((size_t)GATES * IN_DIM * 2);
  _Float16* whh0b  = (_Float16*)take((size_t)GATES * HID * 2);
  _Float16* wih1f  = (_Float16*)take((size_t)GATES * CAT * 2);
  _Float16* whh1f  = (_Float16*)take((size_t)GATES * HID * 2);
  _Float16* wih1b  = (_Float16*)take((size_t)GATES * CAT * 2);
  _Float16* whh1b  = (_Float16*)take((size_t)GATES * HID * 2);
  _Float16* fcw16  = (_Float16*)take((size_t)OUT_DIM * CAT * 2);
  _Float16* xwf    = (_Float16*)take((size_t)M_ROWS * GATES * 2);    // 128 MB
  _Float16* xwb    = (_Float16*)take((size_t)M_ROWS * GATES * 2);    // 128 MB
  _Float16* h0cat  = (_Float16*)take((size_t)M_ROWS * CAT * 2);      // 64 MB
  _Float16* h1cat  = (_Float16*)take((size_t)M_ROWS * CAT * 2);      // 64 MB

  const int THR = 256;
  auto cvt = [&](const float* src, _Float16* dst, int n) {
    f32_to_f16_kernel<<<dim3((n + THR - 1) / THR), dim3(THR), 0, stream>>>(
        src, dst, n);
  };
  // Weight conversion f32 -> f16
  cvt(w_ih_0f, wih0f, GATES * IN_DIM);
  cvt(w_hh_0f, whh0f, GATES * HID);
  cvt(w_ih_0b, wih0b, GATES * IN_DIM);
  cvt(w_hh_0b, whh0b, GATES * HID);
  cvt(w_ih_1f, wih1f, GATES * CAT);
  cvt(w_hh_1f, whh1f, GATES * HID);
  cvt(w_ih_1b, wih1b, GATES * CAT);
  cvt(w_hh_1b, whh1b, GATES * HID);
  cvt(fc_w,    fcw16, OUT_DIM * CAT);

  // Pack input sequence: [B,IN,T] f32 -> [T,B,IN] f16
  {
    int n = T_LEN * BATCH * IN_DIM;
    pack_seq_kernel<<<dim3((n + THR - 1) / THR), dim3(THR), 0, stream>>>(x,
                                                                         seq0);
  }

  dim3 ggrid(M_ROWS / BM, GATES / BN);  // 1024 x 4
  // Layer 0 input projections
  gemm_xw_kernel<<<ggrid, dim3(256), 0, stream>>>(seq0, wih0f, b_0f, xwf,
                                                  M_ROWS, IN_DIM);
  gemm_xw_kernel<<<ggrid, dim3(256), 0, stream>>>(seq0, wih0b, b_0b, xwb,
                                                  M_ROWS, IN_DIM);
  // Layer 0 recurrence (fwd + bwd concurrently)
  lstm_layer_kernel<<<dim3(2), dim3(512), 0, stream>>>(xwf, xwb, whh0f, whh0b,
                                                       h0cat);
  // Layer 1 input projections (reuse xw buffers)
  gemm_xw_kernel<<<ggrid, dim3(256), 0, stream>>>(h0cat, wih1f, b_1f, xwf,
                                                  M_ROWS, CAT);
  gemm_xw_kernel<<<ggrid, dim3(256), 0, stream>>>(h0cat, wih1b, b_1b, xwb,
                                                  M_ROWS, CAT);
  // Layer 1 recurrence
  lstm_layer_kernel<<<dim3(2), dim3(512), 0, stream>>>(xwf, xwb, whh1f, whh1b,
                                                       h1cat);
  // Final FC on last timestep
  fc_kernel<<<dim3(1), dim3(256), 0, stream>>>(
      h1cat + (size_t)(T_LEN - 1) * BATCH * CAT, fcw16, fc_b, out);
}
